// DictPlenoxels_78245714199141
// MI455X (gfx1250) — matrix operand order repaired and checked by reading, not compiled, not run on hardware
//
#include <hip/hip_runtime.h>
#include <hip/hip_bf16.h>
#include <math.h>

typedef __attribute__((ext_vector_type(16))) _Float16 v16h;
typedef __attribute__((ext_vector_type(8)))  _Float16 h8;
typedef __attribute__((ext_vector_type(8)))  float    v8f;

#define RADIUS_F   1.3f
#define COARSE     32
#define FINE       2
#define NUM_ATOMS  32
#define DATA_DIM   28      // SH_DIM*3 + 1
#define BATCH      512
#define NSAMP      383     // N_INT - 1
#define KDIM       256     // 8 fine cells * 32 atoms
#define STEP_F     (2.0f * RADIUS_F / (COARSE * FINE * 2))   // 0.0203125
#define FINE_VLEN  (2.0f * RADIUS_F / (COARSE * FINE))       // 0.040625
#define ATOM_BYTES (8 * NUM_ATOMS * DATA_DIM * 4)            // 28672 B

// ---------------------------------------------------------------------------
// Kernel 1: build per-point coefficient vectors (trilinear gather over 8
// corners into a K=256 vector) and multiply by the stacked atoms dictionary
// via v_wmma_f32_16x16x32_f16.  One block = 64 threads = 2 waves = 64 points.
// Atoms dictionary is staged into LDS with global_load_async_to_lds_b128.
// ---------------------------------------------------------------------------
__launch_bounds__(64)
__global__ void DictPlenoxels_interp_wmma(const float* __restrict__ rays_o,
                                          const float* __restrict__ rays_d,
                                          const float* __restrict__ grid,
                                          const float* __restrict__ atoms,
                                          float* __restrict__ data)
{
    // B fragments in WMMA lane order: [kblock(8)][ntile(2)][lane(32)][h(16)]
    __shared__ __align__(32) _Float16 ldsBf[8 * 2 * 32 * 16];     // 16 KB
    // Per-point coefficient rows; its storage doubles as the raw-atoms
    // staging buffer (28 KB < 32 KB) before coefficients are built.
    __shared__ __align__(16) _Float16 ldsCoef[2][32][KDIM];       // 32 KB

    const int tid  = threadIdx.x;
    const int wave = tid >> 5;
    const int lane = tid & 31;

    // ---- Stage raw atoms (f32) into LDS via async DMA ---------------------
    float* ldsRaw = (float*)&ldsCoef[0][0][0];
    {
        unsigned ldsBase = (unsigned)(uintptr_t)ldsRaw;   // low 32 bits of a
                                                          // generic LDS addr
                                                          // == LDS offset
        #pragma unroll
        for (int j = 0; j < ATOM_BYTES / (64 * 16); ++j) {  // 28 iters
            const unsigned byteOff = (unsigned)(tid + j * 64) * 16u;
            const unsigned dstOff  = ldsBase + byteOff;
            asm volatile("global_load_async_to_lds_b128 %0, %1, %2 offset:0"
                         :: "v"(dstOff), "v"(byteOff), "s"(atoms)
                         : "memory");
        }
        asm volatile("s_wait_asynccnt 0x0" ::: "memory");
    }
    __syncthreads();

    // ---- Convert staged atoms into pre-swizzled f16 B fragments -----------
    // B (32x16 f16) lane layout: lanes 0-15 hold K=0..15, lanes 16-31 hold
    // K=16..31; half h maps to K = kbB + h (contiguous).
    for (int rid = tid; rid < 512; rid += 64) {
        const int kblk = rid >> 6;
        const int nt   = (rid >> 5) & 1;
        const int l    = rid & 31;
        const int n    = (l & 15) + nt * 16;
        const int kbeg = kblk * 32 + ((l < 16) ? 0 : 16);
        v16h frag;
        #pragma unroll
        for (int h = 0; h < 16; ++h) {
            const int k = kbeg + h;                       // k = f*32 + a
            frag[h] = (n < DATA_DIM) ? (_Float16)ldsRaw[k * DATA_DIM + n]
                                     : (_Float16)0.0f;
        }
        *(v16h*)&ldsBf[rid * 16] = frag;
    }
    __syncthreads();   // raw buffer free; coef region reusable now

    // ---- Per-point ray sample -------------------------------------------
    const int P   = blockIdx.x * 64 + wave * 32 + lane;     // global point id
    const int ray = P / NSAMP;
    const int s   = P - ray * NSAMP;

    const float ox = rays_o[ray * 3 + 0], oy = rays_o[ray * 3 + 1], oz = rays_o[ray * 3 + 2];
    const float dx = rays_d[ray * 3 + 0], dy = rays_d[ray * 3 + 1], dz = rays_d[ray * 3 + 2];

    const float sx = fminf((RADIUS_F - ox) / dx, (-RADIUS_F - ox) / dx);
    const float sy = fminf((RADIUS_F - oy) / dy, (-RADIUS_F - oy) / dy);
    const float sz = fminf((RADIUS_F - oz) / dz, (-RADIUS_F - oz) / dz);
    const float start = fmaxf(sx, fmaxf(sy, sz));

    const float t  = start + (float)s * STEP_F;
    const float px = ox + t * dx, py = oy + t * dy, pz = oz + t * dz;
    const bool mask = (px > -RADIUS_F) && (px < RADIUS_F) &&
                      (py > -RADIUS_F) && (py < RADIUS_F) &&
                      (pz > -RADIUS_F) && (pz < RADIUS_F);

    const float qx = (px + RADIUS_F) / FINE_VLEN;
    const float qy = (py + RADIUS_F) / FINE_VLEN;
    const float qz = (pz + RADIUS_F) / FINE_VLEN;

    // Zero my coefficient row (vectorized).
    {
        h8* crow8 = (h8*)&ldsCoef[wave][lane][0];
        #pragma unroll
        for (int c = 0; c < KDIM / 8; ++c) crow8[c] = (h8)(_Float16)0.0f;
    }

    // ---- Corner accumulation (packed f16 RMW in LDS) ----------------------
    #pragma unroll
    for (int n = 0; n < 8; ++n) {
        const float offx = (n & 4) ? 0.5f : -0.5f;
        const float offy = (n & 2) ? 0.5f : -0.5f;
        const float offz = (n & 1) ? 0.5f : -0.5f;
        const float fx = fminf(fmaxf(floorf(qx + offx), 0.0f), 63.0f);
        const float fy = fminf(fmaxf(floorf(qy + offy), 0.0f), 63.0f);
        const float fz = fminf(fmaxf(floorf(qz + offz), 0.0f), 63.0f);
        float wgt = (1.0f - fabsf(qx - (fx + 0.5f))) *
                    (1.0f - fabsf(qy - (fy + 0.5f))) *
                    (1.0f - fabsf(qz - (fz + 0.5f)));
        if (!mask) wgt = 0.0f;                 // masked points never contribute
        if (wgt != 0.0f) {
            const int ix = (int)fx, iy = (int)fy, iz = (int)fz;
            const int cflat = ((ix >> 1) * COARSE + (iy >> 1)) * COARSE + (iz >> 1);
            const int fflat = ((ix & 1) * FINE + (iy & 1)) * FINE + (iz & 1);
            const float4* g4 = (const float4*)(grid + (size_t)cflat * NUM_ATOMS);
            h8* crow8 = (h8*)&ldsCoef[wave][lane][fflat * NUM_ATOMS];
            #pragma unroll
            for (int c = 0; c < 4; ++c) {
                const float4 ga = g4[c * 2 + 0];
                const float4 gb = g4[c * 2 + 1];
                h8 add;
                add[0] = (_Float16)(wgt * ga.x); add[1] = (_Float16)(wgt * ga.y);
                add[2] = (_Float16)(wgt * ga.z); add[3] = (_Float16)(wgt * ga.w);
                add[4] = (_Float16)(wgt * gb.x); add[5] = (_Float16)(wgt * gb.y);
                add[6] = (_Float16)(wgt * gb.z); add[7] = (_Float16)(wgt * gb.w);
                crow8[c] = crow8[c] + add;     // ds b128 RMW + v_pk_add_f16
            }
        }
    }
    __syncthreads();

    // ---- WMMA GEMM: coef[16 x 256] @ atoms[256 x 32] ----------------------
    // A (16x32 f16) lane layout: lanes 0-15 -> K = kbA+0..15
    // with kbA = 0; lanes 16-31 -> kbA = 16.  One contiguous 16-half run.
    const int kbA  = (lane < 16) ? 0 : 16;
    const int mrow = lane & 15;
    const int n0   = lane & 15;

    #pragma unroll
    for (int mt = 0; mt < 2; ++mt) {
        v8f c0 = {}; v8f c1 = {};
        const _Float16* arow = &ldsCoef[wave][mt * 16 + mrow][0];
        #pragma unroll
        for (int kblk = 0; kblk < 8; ++kblk) {
            const int kk = kblk * 32;
            const h8 alo = *(const h8*)&arow[kk + kbA];
            const h8 ahi = *(const h8*)&arow[kk + kbA + 8];
            const v16h a = __builtin_shufflevector(alo, ahi,
                0, 1, 2, 3, 4, 5, 6, 7, 8, 9, 10, 11, 12, 13, 14, 15);
            const v16h b0 = *(const v16h*)&ldsBf[((kblk * 2 + 0) * 32 + lane) * 16];
            const v16h b1 = *(const v16h*)&ldsBf[((kblk * 2 + 1) * 32 + lane) * 16];
            c0 = __builtin_amdgcn_wmma_f32_16x16x32_f16(false, a, false, b0, (short)0, c0, false, false);
            c1 = __builtin_amdgcn_wmma_f32_16x16x32_f16(false, a, false, b1, (short)0, c1, false, false);
        }
        // D layout: lanes 0-15: N=lane, M=r ; lanes 16-31: N=lane-16, M=r+8
        const int Pbase = blockIdx.x * 64 + wave * 32 + mt * 16;
        #pragma unroll
        for (int r = 0; r < 8; ++r) {
            const int M = r + ((lane < 16) ? 0 : 8);
            float* drow = data + (size_t)(Pbase + M) * DATA_DIM;
            drow[n0] = c0[r];
            if (n0 + 16 < DATA_DIM) drow[n0 + 16] = c1[r];
        }
    }
}

// ---------------------------------------------------------------------------
// Kernel 2: per-ray volume rendering. One wave32 per ray; shuffle-based
// inclusive product scan for the transmittance cumprod with a chunk carry.
// ---------------------------------------------------------------------------
__launch_bounds__(32)
__global__ void DictPlenoxels_render(const float* __restrict__ rays_o,
                                     const float* __restrict__ rays_d,
                                     const float* __restrict__ data,
                                     float* __restrict__ out_rgb,
                                     float* __restrict__ out_alpha,
                                     float* __restrict__ out_depth,
                                     float* __restrict__ out_loss)
{
    const int ray  = blockIdx.x;
    const int lane = threadIdx.x;

    const float ox = rays_o[ray * 3 + 0], oy = rays_o[ray * 3 + 1], oz = rays_o[ray * 3 + 2];
    const float dx = rays_d[ray * 3 + 0], dy = rays_d[ray * 3 + 1], dz = rays_d[ray * 3 + 2];
    const float dn = sqrtf(dx * dx + dy * dy + dz * dz);

    const float sx = fminf((RADIUS_F - ox) / dx, (-RADIUS_F - ox) / dx);
    const float sy = fminf((RADIUS_F - oy) / dy, (-RADIUS_F - oy) / dy);
    const float sz = fminf((RADIUS_F - oz) / dz, (-RADIUS_F - oz) / dz);
    const float start = fmaxf(sx, fmaxf(sy, sz));
    const float dist  = STEP_F * dn;

    // SH degree-2 basis on normalized direction
    const float nx = dx / dn, ny = dy / dn, nz = dz / dn;
    float sh[9];
    sh[0] =  0.28209479177387814f;
    sh[1] = -0.48860251190291987f * ny;
    sh[2] =  0.48860251190291987f * nz;
    sh[3] = -0.48860251190291987f * nx;
    sh[4] =  1.0925484305920792f * nx * ny;
    sh[5] = -1.0925484305920792f * ny * nz;
    sh[6] =  0.31539156525252005f * (2.0f * nz * nz - nx * nx - ny * ny);
    sh[7] = -1.0925484305920792f * nx * nz;
    sh[8] =  0.5462742152960396f * (nx * nx - ny * ny);

    float carry = 1.0f;
    float r0a = 0.0f, r1a = 0.0f, r2a = 0.0f, sabsl = 0.0f, dep = 0.0f;

    for (int chunk = 0; chunk < 12; ++chunk) {
        const int s = chunk * 32 + lane;
        const bool valid = (s < NSAMP);
        float alpha = 0.0f, t = 0.0f, v[28];
        bool mask = false;
        if (valid) {
            t = start + (float)s * STEP_F;
            const float px = ox + t * dx, py = oy + t * dy, pz = oz + t * dz;
            mask = (px > -RADIUS_F) && (px < RADIUS_F) &&
                   (py > -RADIUS_F) && (py < RADIUS_F) &&
                   (pz > -RADIUS_F) && (pz < RADIUS_F);
            const float4* row = (const float4*)(data + (size_t)(ray * NSAMP + s) * DATA_DIM);
            #pragma unroll
            for (int i = 0; i < 7; ++i) {
                const float4 g = row[i];
                v[i * 4 + 0] = g.x; v[i * 4 + 1] = g.y;
                v[i * 4 + 2] = g.z; v[i * 4 + 3] = g.w;
            }
            const float sigma = mask ? fmaxf(v[27], 0.0f) : 0.0f;
            alpha = 1.0f - expf(-sigma * dist);
        }
        // inclusive product scan of q = 1 - alpha + 1e-10 across the wave
        const float q = 1.0f - alpha + 1e-10f;
        float incl = q;
        #pragma unroll
        for (int off = 1; off < 32; off <<= 1) {
            const float u = __shfl_up(incl, off, 32);
            if (lane >= off) incl *= u;
        }
        float excl = __shfl_up(incl, 1, 32);
        if (lane == 0) excl = 1.0f;
        const float T    = carry * excl;
        const float absl = alpha * T;

        if (valid) {
            out_alpha[ray * NSAMP + s] = alpha;
            float c0 = 0.0f, c1 = 0.0f, c2 = 0.0f;
            if (mask) {
                #pragma unroll
                for (int k = 0; k < 9; ++k) {
                    c0 += sh[k] * v[k];
                    c1 += sh[k] * v[9 + k];
                    c2 += sh[k] * v[18 + k];
                }
            }
            r0a += absl / (1.0f + expf(-c0));
            r1a += absl / (1.0f + expf(-c1));
            r2a += absl / (1.0f + expf(-c2));
            sabsl += absl;
            dep   += absl * t;
        }
        carry *= __shfl(incl, 31, 32);
    }

    #pragma unroll
    for (int off = 16; off; off >>= 1) {
        r0a   += __shfl_xor(r0a, off, 32);
        r1a   += __shfl_xor(r1a, off, 32);
        r2a   += __shfl_xor(r2a, off, 32);
        sabsl += __shfl_xor(sabsl, off, 32);
        dep   += __shfl_xor(dep, off, 32);
    }
    if (lane == 0) {
        const float bg = 1.0f - sabsl;
        out_rgb[ray * 3 + 0] = r0a + bg;
        out_rgb[ray * 3 + 1] = r1a + bg;
        out_rgb[ray * 3 + 2] = r2a + bg;
        out_depth[ray] = dep;
        if (ray == 0) out_loss[0] = 0.0f;
    }
}

// ---------------------------------------------------------------------------
extern "C" void kernel_launch(void* const* d_in, const int* in_sizes, int n_in,
                              void* d_out, int out_size, void* d_ws, size_t ws_size,
                              hipStream_t stream) {
    (void)in_sizes; (void)n_in; (void)out_size; (void)ws_size;
    const float* rays_o = (const float*)d_in[0];
    const float* rays_d = (const float*)d_in[1];
    const float* grid   = (const float*)d_in[2];
    const float* atoms  = (const float*)d_in[3];

    float* data = (float*)d_ws;                 // [BATCH*NSAMP, 28] scratch, ~22 MB
    float* out       = (float*)d_out;
    float* out_rgb   = out;                      // [512,3]
    float* out_alpha = out + BATCH * 3;          // [512,383]
    float* out_depth = out_alpha + BATCH * NSAMP;// [512]
    float* out_loss  = out_depth + BATCH;        // [1]

    const int npts = BATCH * NSAMP;              // 196096 = 3064 * 64
    DictPlenoxels_interp_wmma<<<npts / 64, 64, 0, stream>>>(rays_o, rays_d, grid, atoms, data);
    DictPlenoxels_render<<<BATCH, 32, 0, stream>>>(rays_o, rays_d, data,
                                                   out_rgb, out_alpha, out_depth, out_loss);
}
